// RNN_35029753266190
// MI455X (gfx1250) — matrix operand order repaired and checked by reading, not compiled, not run on hardware
//
#include <hip/hip_runtime.h>
#include <hip/hip_bf16.h>

typedef __attribute__((ext_vector_type(16))) _Float16 v16h;
typedef __attribute__((ext_vector_type(8)))  _Float16 v8h;
typedef __attribute__((ext_vector_type(2)))  _Float16 v2h;
typedef __attribute__((ext_vector_type(8)))  float    v8f;
typedef __attribute__((ext_vector_type(4)))  float    v4f;

#define BATCH   16384
#define TSTEPS  28
#define INF     28
#define HID     84
#define KP      96      // hidden K padded to 3 chunks of 32
#define G4      336     // 4*H
#define NT      21      // 336/16 gate-column tiles
#define GP      20      // gate-scratch pitch (floats): 80B rows, 16B aligned
#define PAIRS   4       // 2 waves cooperate per 16-row batch tile
#define ROWS_PER_BLOCK (PAIRS * 16)   // 64
#define LOG2E   1.4426950408889634f

// ---------------- fragment loaders (ISA 7.12.2 wave32 layouts) ----------------
// A 16x32 f16: lane l -> M = l%16; halves[0..7]=K (l/16)*8+0..7 ; halves[8..15]=K 16+(l/16)*8+0..7
template<int PITCH>
__device__ __attribute__((always_inline))
inline v16h load_fragA(const _Float16* base, int lane, int kc) {
  const int half = lane >> 4, m = lane & 15;
  const _Float16* p = base + m * PITCH + kc * 32 + half * 8;
  const v8h lo = *(const v8h*)(p);
  const v8h hi = *(const v8h*)(p + 16);
  v16h r;
#pragma unroll
  for (int i = 0; i < 8; ++i) { r[i] = lo[i]; r[i + 8] = hi[i]; }
  return r;
}

// B 32x16 f16: lane l -> N = l%16; halves[0..7]=K (l/16)*8+0..7 ; halves[8..15]=K 16+(l/16)*8+0..7
// Weights stored row-major [column n*4+g][K], so B[k][col] = W[col][k].
template<int PITCH>
__device__ __attribute__((always_inline))
inline v16h load_fragB(const _Float16* wlds, int lane, int rowBase, int kc) {
  const int half = lane >> 4, n = lane & 15;
  const _Float16* p = wlds + (rowBase + n) * PITCH + kc * 32 + half * 8;
  const v8h lo = *(const v8h*)(p);
  const v8h hi = *(const v8h*)(p + 16);
  v16h r;
#pragma unroll
  for (int i = 0; i < 8; ++i) { r[i] = lo[i]; r[i + 8] = hi[i]; }
  return r;
}

// ---------------- branch-free fast activations (v_exp_f32 / v_rcp_f32) -------
__device__ __attribute__((always_inline)) inline float sigf(float v) {
  return __builtin_amdgcn_rcpf(1.0f + __builtin_amdgcn_exp2f(-LOG2E * v));
}
__device__ __attribute__((always_inline)) inline float tanhfast(float v) {
  float vc = fminf(fmaxf(v, -40.0f), 40.0f);
  float e  = __builtin_amdgcn_exp2f(-2.0f * LOG2E * vc);
  return (1.0f - e) * __builtin_amdgcn_rcpf(1.0f + e);
}

// Half of one LSTM layer step: this wave handles gate tiles [TBASE, TBASE+TCNT).
// Gate columns are gate-interleaved: col = unit*4 + {i,f,g,o}.
template<int NICH, int PITCHI, int TBASE, int TCNT>
__device__ __attribute__((always_inline))
inline void lstm_half_step(const _Float16* Wi, const _Float16* Wh, const float* bias,
                           const v16h (&aIn)[NICH], const v16h (&aH)[3],
                           float (&c)[22], _Float16* hOut, float* gsc, int lane) {
  const int half = lane >> 4, nl = lane & 15;
  const int e0  = lane * 2;
  const int row = e0 >> 2;      // 0..15 (each row covered by 2 lanes)
  const int ul0 = e0 & 3;       // 0 or 2: this lane owns units ul0, ul0+1 of each tile
#pragma unroll
  for (int ti = 0; ti < TCNT; ++ti) {
    const int tile = TBASE + ti;
    v8f acc = {};
#pragma unroll
    for (int kc = 0; kc < NICH; ++kc) {
      v16h b = load_fragB<PITCHI>(Wi, lane, tile * 16, kc);
      acc = __builtin_amdgcn_wmma_f32_16x16x32_f16(false, aIn[kc], false, b,
                                                   (short)0, acc, false, false);
    }
#pragma unroll
    for (int kc = 0; kc < 3; ++kc) {
      v16h b = load_fragB<KP>(Wh, lane, tile * 16, kc);
      acc = __builtin_amdgcn_wmma_f32_16x16x32_f16(false, aH[kc], false, b,
                                                   (short)0, acc, false, false);
    }
    // D layout: lane l, vgpr j -> (M = j + (l/16)*8, N = tile*16 + l%16)
#pragma unroll
    for (int j = 0; j < 8; ++j)
      gsc[(j + half * 8) * GP + nl] = acc[j];
    // vectorized activations: 8 consecutive gates = 2 units (i,f,g,o each)
    const float* gp = gsc + row * GP + ul0 * 4;
    const v4f ga = *(const v4f*)(gp);
    const v4f gb = *(const v4f*)(gp + 4);
    const int cb = tile * 16 + ul0 * 4;
    const v4f ba = *(const v4f*)(bias + cb);
    const v4f bb = *(const v4f*)(bias + cb + 4);
    const float i0 = sigf(ga.x + ba.x), f0 = sigf(ga.y + ba.y);
    const float g0 = tanhfast(ga.z + ba.z), o0 = sigf(ga.w + ba.w);
    const float i1 = sigf(gb.x + bb.x), f1 = sigf(gb.y + bb.y);
    const float g1 = tanhfast(gb.z + bb.z), o1 = sigf(gb.w + bb.w);
    const float cn0 = f0 * c[ti * 2 + 0] + i0 * g0;
    const float cn1 = f1 * c[ti * 2 + 1] + i1 * g1;
    c[ti * 2 + 0] = cn0;
    c[ti * 2 + 1] = cn1;
    v2h hp;
    hp.x = (_Float16)(o0 * tanhfast(cn0));
    hp.y = (_Float16)(o1 * tanhfast(cn1));
    *(v2h*)(hOut + row * KP + tile * 4 + ul0) = hp;   // packed ds_store_b32
  }
}

extern "C" __global__ void __launch_bounds__(256, 1)
lstm2_wmma_kernel(const float* __restrict__ x,
                  const float* __restrict__ wih0, const float* __restrict__ whh0,
                  const float* __restrict__ bih0, const float* __restrict__ bhh0,
                  const float* __restrict__ wih1, const float* __restrict__ whh1,
                  const float* __restrict__ bih1, const float* __restrict__ bhh1,
                  const float* __restrict__ wout, const float* __restrict__ bout,
                  float* __restrict__ out) {
  // ---- LDS (≈254 KB of the 320 KB WGP pool) ----
  __shared__ alignas(16) _Float16 sWih0[G4 * 32];
  __shared__ alignas(16) _Float16 sWhh0[G4 * KP];
  __shared__ alignas(16) _Float16 sWih1[G4 * KP];
  __shared__ alignas(16) _Float16 sWhh1[G4 * KP];
  __shared__ alignas(16) float    sB0[G4];
  __shared__ alignas(16) float    sB1[G4];
  __shared__ alignas(16) float    sWout[10 * HID];
  __shared__ alignas(16) float    sBout[16];
  __shared__ alignas(16) _Float16 sH1[PAIRS][16 * KP];
  __shared__ alignas(16) _Float16 sH2[PAIRS][16 * KP];
  __shared__ alignas(16) _Float16 sX [PAIRS][16 * 32];
  __shared__ alignas(16) float    sG [PAIRS * 2][16 * GP];

  const int tid  = threadIdx.x;
  const int wave = tid >> 5;
  const int lane = tid & 31;
  const int pair = wave >> 1;   // which 16-row batch tile
  const int sub  = wave & 1;    // which half of the 21 gate tiles

  // ---- cooperative weight load, gate-interleaved: LDS row r = n*4+g <- global row g*84+n ----
  for (int r = tid; r < G4; r += 256) {
    const int n = r >> 2, g = r & 3;
    const int grow = g * HID + n;
    const float* s0 = wih0 + (size_t)grow * INF;
    for (int k = 0; k < 32; ++k)
      sWih0[r * 32 + k] = (k < INF) ? (_Float16)s0[k] : (_Float16)0.0f;
    const float* sh0 = whh0 + (size_t)grow * HID;
    const float* si1 = wih1 + (size_t)grow * HID;
    const float* sh1 = whh1 + (size_t)grow * HID;
    for (int k = 0; k < KP; ++k) {
      const bool ok = (k < HID);
      sWhh0[r * KP + k] = ok ? (_Float16)sh0[k] : (_Float16)0.0f;
      sWih1[r * KP + k] = ok ? (_Float16)si1[k] : (_Float16)0.0f;
      sWhh1[r * KP + k] = ok ? (_Float16)sh1[k] : (_Float16)0.0f;
    }
    sB0[r] = bih0[grow] + bhh0[grow];
    sB1[r] = bih1[grow] + bhh1[grow];
  }
  for (int i = tid; i < 10 * HID; i += 256) sWout[i] = wout[i];
  if (tid < 10) sBout[tid] = bout[tid];

  // ---- per-pair state init (both partner waves write identical values) ----
  _Float16* h1  = &sH1[pair][0];
  _Float16* h2  = &sH2[pair][0];
  _Float16* xw  = &sX[pair][0];
  float*    gsc = &sG[wave][0];
  for (int i = lane; i < 16 * KP; i += 32) { h1[i] = (_Float16)0.0f; h2[i] = (_Float16)0.0f; }
  for (int i = lane; i < 16 * 32; i += 32) xw[i] = (_Float16)0.0f;
  __syncthreads();

  const int bbase = blockIdx.x * ROWS_PER_BLOCK + pair * 16;
  float c0[22], c1[22];
#pragma unroll
  for (int i = 0; i < 22; ++i) { c0[i] = 0.0f; c1[i] = 0.0f; }

  // ---- fused two-layer recurrence; partner waves split the 21 gate tiles 11/10 ----
#pragma unroll 1
  for (int t = 0; t < TSTEPS; ++t) {
    // stage x_t tile (both partner waves write identical data -> no barrier needed)
    {
      const int m = lane & 15, k0 = (lane >> 4) * 14;
      const float* xp = x + ((size_t)(bbase + m) * TSTEPS + t) * INF + k0;
      if (t + 1 < TSTEPS) __builtin_prefetch(xp + INF, 0, 1);
#pragma unroll
      for (int j = 0; j < 14; ++j)
        xw[m * 32 + k0 + j] = (_Float16)xp[j];
    }
    // A fragments (h1/h2 complete here: guarded by previous iteration's barrier)
    v16h aX[1];
    aX[0] = load_fragA<32>(xw, lane, 0);
    v16h aH1[3], aH2[3];
#pragma unroll
    for (int kc = 0; kc < 3; ++kc) {
      aH1[kc] = load_fragA<KP>(h1, lane, kc);
      aH2[kc] = load_fragA<KP>(h2, lane, kc);
    }
    // layer 0: gates = x_t @ Wih0^T + h1_prev @ Whh0^T + b -> each wave writes its h1 half
    if (sub == 0)
      lstm_half_step<1, 32, 0, 11>(sWih0, sWhh0, sB0, aX, aH1, c0, h1, gsc, lane);
    else
      lstm_half_step<1, 32, 11, 10>(sWih0, sWhh0, sB0, aX, aH1, c0, h1, gsc, lane);
    __syncthreads();                       // h1 complete across partner waves
    v16h aH1n[3];
#pragma unroll
    for (int kc = 0; kc < 3; ++kc) aH1n[kc] = load_fragA<KP>(h1, lane, kc);
    // layer 1: gates = h1 @ Wih1^T + h2_prev @ Whh1^T + b -> each wave writes its h2 half
    if (sub == 0)
      lstm_half_step<3, KP, 0, 11>(sWih1, sWhh1, sB1, aH1n, aH2, c1, h2, gsc, lane);
    else
      lstm_half_step<3, KP, 11, 10>(sWih1, sWhh1, sB1, aH1n, aH2, c1, h2, gsc, lane);
    __syncthreads();                       // h2 complete across partner waves
  }

  // ---- final projection: out = h2_last @ w_out^T + b_out (one wave per pair) ----
  if (sub == 0) {
#pragma unroll
    for (int j = 0; j < 5; ++j) {
      const int idx = lane * 5 + j;      // 160 = 16 rows x 10 outputs per pair
      const int m = idx / 10, o = idx % 10;
      float acc = sBout[o];
#pragma unroll 4
      for (int k = 0; k < HID; ++k)
        acc += (float)h2[m * KP + k] * sWout[o * HID + k];
      out[(size_t)(bbase + m) * 10 + o] = acc;
    }
  }
}

extern "C" void kernel_launch(void* const* d_in, const int* in_sizes, int n_in,
                              void* d_out, int out_size, void* d_ws, size_t ws_size,
                              hipStream_t stream) {
  (void)in_sizes; (void)n_in; (void)d_ws; (void)ws_size; (void)out_size;
  const float* x    = (const float*)d_in[0];
  const float* wih0 = (const float*)d_in[1];
  const float* whh0 = (const float*)d_in[2];
  const float* bih0 = (const float*)d_in[3];
  const float* bhh0 = (const float*)d_in[4];
  const float* wih1 = (const float*)d_in[5];
  const float* whh1 = (const float*)d_in[6];
  const float* bih1 = (const float*)d_in[7];
  const float* bhh1 = (const float*)d_in[8];
  const float* wout = (const float*)d_in[9];
  const float* bout = (const float*)d_in[10];
  float* out = (float*)d_out;

  dim3 grid(BATCH / ROWS_PER_BLOCK);  // 256 blocks -> covers up to 256 WGPs
  dim3 block(256);                    // 8 waves = 4 pairs; 2 waves per batch tile
  hipLaunchKernelGGL(lstm2_wmma_kernel, grid, block, 0, stream,
                     x, wih0, whh0, bih0, bhh0, wih1, whh1, bih1, bhh1,
                     wout, bout, out);
}
